// KalmanNetNN_3178275799082
// MI455X (gfx1250) — matrix operand order repaired and checked by reading, not compiled, not run on hardware
//
#include <hip/hip_runtime.h>
#include <hip/hip_bf16.h>

typedef __attribute__((ext_vector_type(8)))  _Float16 v8h;
typedef __attribute__((ext_vector_type(16))) _Float16 v16h;
typedef __attribute__((ext_vector_type(8)))  float    v8f;
typedef __attribute__((ext_vector_type(4)))  unsigned u32x4;
typedef __attribute__((ext_vector_type(4)))  int      i32x4;
typedef __attribute__((ext_vector_type(8)))  int      i32x8;

#define NTHREADS 128
#define WAVES 4
#define ROWS_PER_BLOCK (WAVES * 16)

#define HT_TILES 57            // ceil(1800/32)
#define REC_BYTES 8192         // packed weight tile record: Wa(4096) Wb(2048) b2a(128) pad

// workspace layout (all produced by prep kernel)
#define WSOFF_W 0              // resident weights, f16, byte-exact Smem image (76608 B)
#define WSB_W   76608
#define WSOFF_B 76608          // biases fp32, Smem order, padded (2720 B)
#define WSB_B   2720
#define WSOFF_T 81920          // 57 FC2 tile records
#define WS_NEEDED ((size_t)WSOFF_T + (size_t)HT_TILES * REC_BYTES)

// ---------------- kernel arguments ----------------
struct KArgs {
  const float *y, *post, *post_prev, *prior_prev, *y_prev, *hQ0, *hSig0, *hS0;
  const float *FC1_W, *FC1_b, *FC2a_W, *FC2a_b, *FC2b_W, *FC2b_b;
  const float *FC3_W, *FC3_b, *FC4_W, *FC4_b, *FC5_W, *FC5_b;
  const float *FC6_W, *FC6_b, *FC7_W, *FC7_b;
  const float *Q_Wih, *Q_Whh, *Q_bih, *Q_bhh;
  const float *Sig_Wih, *Sig_Whh, *Sig_bih, *Sig_bhh;
  const float *S_Wih, *S_Whh, *S_bih, *S_bhh;
  const char *ws;              // packed workspace (prep kernel output)
  float *out;
  int B;
  int use_ws;                  // 1 if ws holds packed data
};

// ---------------- LDS layout (~153.5 KiB) ----------------
// NOTE: the 12 weight arrays below are a contiguous 76,608-byte image that the
// prep kernel reproduces byte-exactly in ws; phase 0 DMA-copies it in one TDM op.
struct __align__(16) Smem {
  _Float16 W5[30 * 32];       // FC5_W  (K=6  -> 32)
  _Float16 W6[30 * 32];       // FC6_W
  _Float16 W7[30 * 32];       // FC7_W
  _Float16 WQih[108 * 32];    // Q_Wih   (K=30 -> 32)
  _Float16 WQhh[108 * 64];    // Q_Whh   (K=36 -> 64)
  _Float16 WSigih[108 * 96];  // Sig_Wih (K=66 -> 96)
  _Float16 WSighh[108 * 64];  // Sig_Whh (K=36 -> 64)
  _Float16 W1[9 * 64];        // FC1_W   (K=36 -> 64)
  _Float16 WSih[27 * 64];     // S_Wih   (K=39 -> 64)
  _Float16 WShh[27 * 32];     // S_Whh   (K=9  -> 32)
  _Float16 W3[36 * 32];       // FC3_W   (K=27 -> 32)
  _Float16 W4[36 * 96];       // FC4_W   (K=72 -> 96)
  char stream[2][REC_BYTES];  // double-buffered FC2a/FC2b tile records
  _Float16 arena[WAVES][7680];// per-wave activations (f16)
  // contiguous fp32 bias image (675 floats + pad), async-DMA'd from ws
  float b1[9], b2b[18], b3[36], b4[36], b5[30], b6[30], b7[30];
  float bQi[108], bQh[108], bSi[108], bSh[108], bSSi[27], bSSh[27];
  float bpad[8];
};

// Per-wave arena offsets (halves).  H2A overlays U (dead after FC5).
#define OFF_U     0     // [16][32] fw_update_diff   (later: h2 tile A staging)
#define OFF_V     512   // [16][32] fw_evol_diff
#define OFF_OD    1024  // [16][32] obs_diff(0..2) | obs_innov_diff(3..5)
#define OFF_F5    1536  // [16][32] out_FC5
#define OFF_F3IN  2048  // [16][32] hS(0..8) | out_FC2(9..26)
#define OFF_SIN   2560  // [16][64] out_FC1(0..8) | out_FC7(9..38)
#define OFF_FC2IN 3584  // [16][64] hSig(0..35) | hS(36..44)
#define OFF_SIGIN 4608  // [16][96] hQ(0..35) | out_FC6(36..65)
#define OFF_FC4IN 6144  // [16][96] hSig(0..35) | out_FC3(36..71)
#define ARENA_H   7680

// Generic LDS pointer -> wave-relative LDS byte offset (addr[31:0] per flat aperture).
__device__ __forceinline__ unsigned lds_off_of(const void *p) {
  return (unsigned)(unsigned long long)p;
}

// ---------------- async global->LDS DMA (ASYNCcnt; ISA 10.x/15.18) ----------------
__device__ __forceinline__ void issue_async_b128(const char *src, unsigned ldsOff,
                                                 int n16, int tid) {
  for (int i = tid; i < n16; i += NTHREADS) {
    unsigned idx = (unsigned)i * 16u;
    asm volatile("global_load_async_to_lds_b128 %0, %1, %2"
                 :: "v"(ldsOff + idx), "v"(idx), "s"(src)
                 : "memory");
  }
}
__device__ __forceinline__ void wait_async0() {
  asm volatile("s_wait_asynccnt 0x0" ::: "memory");
}

// ---------------- TDM: 1-D contiguous tensor_load_to_lds (TENSORcnt; ISA ch.8) ----
// Copies nelems elements of (1<<elem_log2) bytes from gsrc to LDS offset ldsOff.
// D# group0: count=1, lds_addr, global_addr[56:0], type=2.
// D# group1: data_size, tensor_dim0=tile_dim0=nelems, tensor_dim1=1, strides=nelems.
__device__ __forceinline__ void tdm_load_1d(const void *gsrc, unsigned ldsOff,
                                            unsigned nelems, unsigned elem_log2) {
  unsigned long long ga = (unsigned long long)gsrc;
  u32x4 g0;
  g0[0] = 1u;                                           // count=1 (valid user D#)
  g0[1] = ldsOff;                                       // lds_addr
  g0[2] = (unsigned)(ga & 0xffffffffu);                 // global_addr[31:0]
  g0[3] = (unsigned)((ga >> 32) & 0x01ffffffu) | (2u << 30);  // addr[56:32]|type=2
  i32x8 g1;
  g1[0] = (int)(elem_log2 << 16);                       // wg_mask=0, data_size
  g1[1] = (int)((nelems & 0xffffu) << 16);              // tensor_dim0[15:0]
  g1[2] = (int)((nelems >> 16) | (1u << 16));           // tensor_dim0[31:16]|td1=1
  g1[3] = (int)((nelems & 0xffffu) << 16);              // td1 hi=0 | tile_dim0
  g1[4] = 0;                                            // tile_dim1=0, tile_dim2=0
  g1[5] = (int)nelems;                                  // dim0_stride[31:0]
  g1[6] = (int)((nelems & 0xffffu) << 16);              // d0s hi=0 | d1s[15:0]
  g1[7] = (int)(nelems >> 16);                          // d1s[47:16]
  i32x4 g2;                                             // dims 2/3 benign values
  g2[0] = 1; g2[1] = 1; g2[2] = (int)nelems; g2[3] = 0;
  i32x4 g3;
  g3[0] = (int)nelems; g3[1] = (1 << 16); g3[2] = 0; g3[3] = 0;
#if defined(__clang_major__) && (__clang_major__ >= 23)
  i32x8 g4 = {};
  __builtin_amdgcn_tensor_load_to_lds(g0, g1, g2, g3, g4, 0);
#else
  __builtin_amdgcn_tensor_load_to_lds(g0, g1, g2, g3, 0);
#endif
}
__device__ __forceinline__ void wait_tensor0() {
  __builtin_amdgcn_s_wait_tensorcnt(0);
}

// ---------------- WMMA fragment helpers (ISA 7.12.2 layouts) ----------------
__device__ __forceinline__ v16h ldfragA(const _Float16 *X, int stride, int s, int lane) {
  const _Float16 *p = X + (lane & 15) * stride + s * 32 + ((lane >> 4) << 3);
  v8h lo = *(const v8h *)p;
  v8h hi = *(const v8h *)(p + 16);
  return __builtin_shufflevector(lo, hi, 0,1,2,3,4,5,6,7,8,9,10,11,12,13,14,15);
}
__device__ __forceinline__ v16h ldfragB(const _Float16 *W, int stride, int rows,
                                        int nt, int s, int lane) {
  int n = nt * 16 + (lane & 15);
  if (n >= rows) n = rows - 1;  // harmless clamp; affects unused columns only
  const _Float16 *p = W + n * stride + s * 32 + ((lane >> 4) << 3);
  v8h lo = *(const v8h *)p;
  v8h hi = *(const v8h *)(p + 16);
  return __builtin_shufflevector(lo, hi, 0,1,2,3,4,5,6,7,8,9,10,11,12,13,14,15);
}

__device__ __forceinline__ v8f wmma_gemm(const _Float16 *X, int xs,
                                         const _Float16 *W, int ws, int wrows,
                                         int nt, int ksteps, v8f acc, int lane) {
  for (int s = 0; s < ksteps; ++s) {
    v16h a = ldfragA(X, xs, s, lane);
    v16h b = ldfragB(W, ws, wrows, nt, s, lane);
    acc = __builtin_amdgcn_wmma_f32_16x16x32_f16(false, a, false, b, (short)0, acc,
                                                 false, false);
  }
  return acc;
}

__device__ __forceinline__ v8f bias_init(const float *b, int nreal, int nt, int lane) {
  int n = nt * 16 + (lane & 15);
  float v = (n < nreal) ? b[n] : 0.0f;
  v8f a;
#pragma unroll
  for (int i = 0; i < 8; ++i) a[i] = v;
  return a;
}
__device__ __forceinline__ v8f relu8(v8f a) {
#pragma unroll
  for (int i = 0; i < 8; ++i) a[i] = fmaxf(a[i], 0.0f);
  return a;
}
__device__ __forceinline__ v8f sig8(v8f a) {
#pragma unroll
  for (int i = 0; i < 8; ++i) a[i] = 1.0f / (1.0f + __expf(-a[i]));
  return a;
}
__device__ __forceinline__ v8f tanh8(v8f a) {
#pragma unroll
  for (int i = 0; i < 8; ++i) a[i] = tanhf(a[i]);
  return a;
}

__device__ __forceinline__ void storeC(_Float16 *buf, int stride, int colbase, int nt,
                                       int nreal, v8f c, int lane) {
  int n = nt * 16 + (lane & 15);
  if (n >= nreal) return;
  int mb = (lane >> 4) * 8;
#pragma unroll
  for (int r = 0; r < 8; ++r)
    buf[(mb + r) * stride + colbase + n] = (_Float16)c[r];
}

// ---------------- fused layers ----------------
template <int NT>
__device__ __forceinline__ void fc_layer(const _Float16 *X, int xs, int ks,
                                         const _Float16 *W, int ws, int Nreal,
                                         const float *bias, bool dorelu,
                                         _Float16 *o1, int o1s, int o1c,
                                         _Float16 *o2, int o2s, int o2c, int lane) {
  v8f acc[NT];
#pragma unroll
  for (int t = 0; t < NT; ++t) {
    acc[t] = wmma_gemm(X, xs, W, ws, Nreal, t, ks, bias_init(bias, Nreal, t, lane), lane);
    if (dorelu) acc[t] = relu8(acc[t]);
  }
  asm volatile("" ::: "memory");
#pragma unroll
  for (int t = 0; t < NT; ++t) {
    storeC(o1, o1s, o1c, t, Nreal, acc[t], lane);
    if (o2) storeC(o2, o2s, o2c, t, Nreal, acc[t], lane);
  }
  asm volatile("" ::: "memory");
}

template <int NT>
__device__ __forceinline__ void gru(const _Float16 *Xi, int xis, int ksI,
                                    const _Float16 *Wih, int wihS,
                                    _Float16 *H, int hs, int ksH,
                                    const _Float16 *Whh, int whhS,
                                    const float *bih, const float *bhh, int Hdim,
                                    _Float16 *out2, int o2s, int o2c, int lane) {
  v8f hnew[NT];
#pragma unroll
  for (int t = 0; t < NT; ++t) {
    v8f ir = wmma_gemm(Xi, xis, Wih,                  wihS, Hdim, t, ksI, bias_init(bih,            Hdim, t, lane), lane);
    v8f iz = wmma_gemm(Xi, xis, Wih + Hdim * wihS,    wihS, Hdim, t, ksI, bias_init(bih + Hdim,     Hdim, t, lane), lane);
    v8f in_= wmma_gemm(Xi, xis, Wih + 2 * Hdim * wihS,wihS, Hdim, t, ksI, bias_init(bih + 2 * Hdim, Hdim, t, lane), lane);
    v8f hr = wmma_gemm(H, hs,  Whh,                   whhS, Hdim, t, ksH, bias_init(bhh,            Hdim, t, lane), lane);
    v8f hz = wmma_gemm(H, hs,  Whh + Hdim * whhS,     whhS, Hdim, t, ksH, bias_init(bhh + Hdim,     Hdim, t, lane), lane);
    v8f hn = wmma_gemm(H, hs,  Whh + 2 * Hdim * whhS, whhS, Hdim, t, ksH, bias_init(bhh + 2 * Hdim, Hdim, t, lane), lane);
    v8f r = sig8(ir + hr);
    v8f z = sig8(iz + hz);
    v8f n = tanh8(in_ + r * hn);
    int nc = t * 16 + (lane & 15);
    int mb = (lane >> 4) * 8;
    v8f hv;
#pragma unroll
    for (int q = 0; q < 8; ++q) {
      float hold = (nc < Hdim) ? (float)H[(mb + q) * hs + nc] : 0.0f;
      hv[q] = (1.0f - z[q]) * n[q] + z[q] * hold;
    }
    hnew[t] = hv;
  }
  asm volatile("" ::: "memory");
#pragma unroll
  for (int t = 0; t < NT; ++t) {
    int nc = t * 16 + (lane & 15);
    if (nc < Hdim) {
      int mb = (lane >> 4) * 8;
#pragma unroll
      for (int q = 0; q < 8; ++q) {
        _Float16 hv = (_Float16)hnew[t][q];
        H[(mb + q) * hs + nc] = hv;
        if (out2) out2[(mb + q) * o2s + o2c + nc] = hv;
      }
    }
  }
  asm volatile("" ::: "memory");
}

// ---------------- scalar staging fallbacks (use_ws == 0) ----------------
__device__ __forceinline__ void stageW(_Float16 *dst, const float *src, int N, int K,
                                       int Kpad, int tid) {
  int tot = N * Kpad;
  for (int i = tid; i < tot; i += NTHREADS) {
    int n = i / Kpad, k = i - n * Kpad;
    dst[i] = (_Float16)((k < K) ? src[n * K + k] : 0.0f);
  }
}
__device__ __forceinline__ void stageV(float *dst, const float *src, int n, int tid) {
  for (int i = tid; i < n; i += NTHREADS) dst[i] = src[i];
}

// ---------------- prep kernel: pack everything into ws ----------------
__device__ __forceinline__ void packW(_Float16 *dst, const float *src, int N, int K,
                                      int Kpad) {
  int tot = N * Kpad;
  for (int i = threadIdx.x; i < tot; i += 256) {
    int n = i / Kpad, k = i - n * Kpad;
    dst[i] = (_Float16)((k < K) ? src[n * K + k] : 0.0f);
  }
}
__device__ __forceinline__ void packB(float *dst, const float *src, int n) {
  for (int i = threadIdx.x; i < n; i += 256) dst[i] = src[i];
}

__global__ void __launch_bounds__(256) pack_all(KArgs g, char *ws) {
  const int t = blockIdx.x;
  if (t < HT_TILES) {
    // FC2 tile record t
    char *rec = ws + WSOFF_T + (size_t)t * REC_BYTES;
    _Float16 *wa = (_Float16 *)rec;            // [32 n][64 k]
    _Float16 *wb = (_Float16 *)(rec + 4096);   // [32 n][32 k]
    float *b2a = (float *)(rec + 6144);        // [32]
    float *pad = (float *)(rec + 6272);
    for (int i = threadIdx.x; i < 32 * 64; i += 256) {
      int n = i >> 6, k = i & 63;
      int gn = t * 32 + n;
      wa[i] = (_Float16)((gn < 1800 && k < 45) ? g.FC2a_W[gn * 45 + k] : 0.0f);
    }
    for (int i = threadIdx.x; i < 32 * 32; i += 256) {
      int n = i >> 5, k = i & 31;
      int gk = t * 32 + k;
      wb[i] = (_Float16)((n < 18 && gk < 1800) ? g.FC2b_W[n * 1800 + gk] : 0.0f);
    }
    for (int i = threadIdx.x; i < 32; i += 256) {
      int gn = t * 32 + i;
      b2a[i] = (gn < 1800) ? g.FC2a_b[gn] : 0.0f;
    }
    for (int i = threadIdx.x; i < 480; i += 256) pad[i] = 0.0f;
  } else {
    // resident weight image (byte-exact Smem order) + bias image
    _Float16 *w = (_Float16 *)(ws + WSOFF_W);
    packW(w + 0,     g.FC5_W,   30, 6, 32);
    packW(w + 960,   g.FC6_W,   30, 6, 32);
    packW(w + 1920,  g.FC7_W,   30, 6, 32);
    packW(w + 2880,  g.Q_Wih,  108, 30, 32);
    packW(w + 6336,  g.Q_Whh,  108, 36, 64);
    packW(w + 13248, g.Sig_Wih,108, 66, 96);
    packW(w + 23616, g.Sig_Whh,108, 36, 64);
    packW(w + 30528, g.FC1_W,    9, 36, 64);
    packW(w + 31104, g.S_Wih,   27, 39, 64);
    packW(w + 32832, g.S_Whh,   27, 9, 32);
    packW(w + 33696, g.FC3_W,   36, 27, 32);
    packW(w + 34848, g.FC4_W,   36, 72, 96);
    float *bd = (float *)(ws + WSOFF_B);
    packB(bd + 0,   g.FC1_b, 9);
    packB(bd + 9,   g.FC2b_b, 18);
    packB(bd + 27,  g.FC3_b, 36);
    packB(bd + 63,  g.FC4_b, 36);
    packB(bd + 99,  g.FC5_b, 30);
    packB(bd + 129, g.FC6_b, 30);
    packB(bd + 159, g.FC7_b, 30);
    packB(bd + 189, g.Q_bih, 108);
    packB(bd + 297, g.Q_bhh, 108);
    packB(bd + 405, g.Sig_bih, 108);
    packB(bd + 513, g.Sig_bhh, 108);
    packB(bd + 621, g.S_bih, 27);
    packB(bd + 648, g.S_bhh, 27);
    for (int i = threadIdx.x; i < 5; i += 256) bd[675 + i] = 0.0f;  // pad to 2720B
  }
}

// ---------------- fused KalmanNet kernel ----------------
__global__ void __launch_bounds__(NTHREADS) knet_fused(KArgs g) {
  __shared__ Smem sm;
  const int tid = threadIdx.x;
  const int lane = tid & 31;
  const int wave = tid >> 5;

  // ---- Phase 0: resident weights + biases into LDS, zero activation arena ----
  if (g.use_ws) {
    if (wave == 0)  // one TDM op copies the whole 76,608-byte weight image
      tdm_load_1d(g.ws + WSOFF_W, lds_off_of(&sm.W5[0]), WSB_W / 8, 3);
    issue_async_b128(g.ws + WSOFF_B, lds_off_of(&sm.b1[0]), WSB_B / 16, tid);
  } else {
    stageW(sm.W5, g.FC5_W, 30, 6, 32, tid);
    stageW(sm.W6, g.FC6_W, 30, 6, 32, tid);
    stageW(sm.W7, g.FC7_W, 30, 6, 32, tid);
    stageW(sm.WQih, g.Q_Wih, 108, 30, 32, tid);
    stageW(sm.WQhh, g.Q_Whh, 108, 36, 64, tid);
    stageW(sm.WSigih, g.Sig_Wih, 108, 66, 96, tid);
    stageW(sm.WSighh, g.Sig_Whh, 108, 36, 64, tid);
    stageW(sm.W1, g.FC1_W, 9, 36, 64, tid);
    stageW(sm.WSih, g.S_Wih, 27, 39, 64, tid);
    stageW(sm.WShh, g.S_Whh, 27, 9, 32, tid);
    stageW(sm.W3, g.FC3_W, 36, 27, 32, tid);
    stageW(sm.W4, g.FC4_W, 36, 72, 96, tid);
    stageV(sm.b1, g.FC1_b, 9, tid);
    stageV(sm.b2b, g.FC2b_b, 18, tid);
    stageV(sm.b3, g.FC3_b, 36, tid);
    stageV(sm.b4, g.FC4_b, 36, tid);
    stageV(sm.b5, g.FC5_b, 30, tid);
    stageV(sm.b6, g.FC6_b, 30, tid);
    stageV(sm.b7, g.FC7_b, 30, tid);
    stageV(sm.bQi, g.Q_bih, 108, tid);
    stageV(sm.bQh, g.Q_bhh, 108, tid);
    stageV(sm.bSi, g.Sig_bih, 108, tid);
    stageV(sm.bSh, g.Sig_bhh, 108, tid);
    stageV(sm.bSSi, g.S_bih, 27, tid);
    stageV(sm.bSSh, g.S_bhh, 27, tid);
  }
  {
    v8h zz = {};
    _Float16 *az = &sm.arena[0][0];
    for (int i = tid * 8; i < WAVES * ARENA_H; i += NTHREADS * 8)
      *(v8h *)(az + i) = zz;
  }
  if (g.use_ws) {
    if (wave == 0) wait_tensor0();
    wait_async0();
  }
  __syncthreads();

  _Float16 *AR = &sm.arena[wave][0];
  _Float16 *U = AR + OFF_U, *V = AR + OFF_V, *OD = AR + OFF_OD, *F5 = AR + OFF_F5;
  _Float16 *F3IN = AR + OFF_F3IN, *H2A = AR + OFF_U /*overlay*/, *SIN = AR + OFF_SIN;
  _Float16 *FC2IN = AR + OFF_FC2IN, *SIGIN = AR + OFF_SIGIN, *FC4IN = AR + OFF_FC4IN;

  const int rowbase = blockIdx.x * ROWS_PER_BLOCK + wave * 16;
  const unsigned streamOff0 = lds_off_of(&sm.stream[0][0]);

  // ---- Phase 1: per-row preprocessing (lane m handles batch row rowbase+m) ----
  float dy0 = 0.f, dy1 = 0.f, dy2 = 0.f;
  float pr[6] = {0.f, 0.f, 0.f, 0.f, 0.f, 0.f};
  if (lane < 16) {
    int b = rowbase + lane;
    if (b < g.B) {
#pragma unroll
      for (int i = 0; i < 6; ++i) pr[i] = g.post[b * 6 + i];
      float yv[3];
#pragma unroll
      for (int j = 0; j < 3; ++j) yv[j] = g.y[b * 3 + j] + pr[3 + j];
      float od[3];
#pragma unroll
      for (int j = 0; j < 3; ++j) od[j] = yv[j] - g.y_prev[b * 3 + j];
      float nn = fmaxf(sqrtf(od[0]*od[0] + od[1]*od[1] + od[2]*od[2]), 1e-12f);
#pragma unroll
      for (int j = 0; j < 3; ++j) OD[lane * 32 + j] = (_Float16)(od[j] / nn);
      dy0 = yv[0] - pr[0]; dy1 = yv[1] - pr[1]; dy2 = yv[2] - pr[2];
      nn = fmaxf(sqrtf(dy0*dy0 + dy1*dy1 + dy2*dy2), 1e-12f);
      OD[lane * 32 + 3] = (_Float16)(dy0 / nn);
      OD[lane * 32 + 4] = (_Float16)(dy1 / nn);
      OD[lane * 32 + 5] = (_Float16)(dy2 / nn);
      float fe[6], fu[6];
      float se = 0.f, su = 0.f;
#pragma unroll
      for (int i = 0; i < 6; ++i) {
        fe[i] = pr[i] - g.post_prev[b * 6 + i];  se += fe[i] * fe[i];
        fu[i] = pr[i] - g.prior_prev[b * 6 + i]; su += fu[i] * fu[i];
      }
      se = fmaxf(sqrtf(se), 1e-12f); su = fmaxf(sqrtf(su), 1e-12f);
#pragma unroll
      for (int i = 0; i < 6; ++i) {
        V[lane * 32 + i] = (_Float16)(fe[i] / se);
        U[lane * 32 + i] = (_Float16)(fu[i] / su);
      }
#pragma unroll
      for (int j = 0; j < 36; ++j) SIGIN[lane * 96 + j] = (_Float16)g.hQ0[b * 36 + j];
#pragma unroll
      for (int j = 0; j < 36; ++j) {
        _Float16 v = (_Float16)g.hSig0[b * 36 + j];
        FC2IN[lane * 64 + j] = v;
        FC4IN[lane * 96 + j] = v;
      }
#pragma unroll
      for (int j = 0; j < 9; ++j) {
        _Float16 v = (_Float16)g.hS0[b * 9 + j];
        F3IN[lane * 32 + j] = v;
        FC2IN[lane * 64 + 36 + j] = v;
      }
    }
  }
  asm volatile("" ::: "memory");

  // ---- FC5/FC6/FC7: state-independent, identical across both steps ----
  fc_layer<2>(U, 32, 1, sm.W5, 32, 30, sm.b5, true, F5, 32, 0, nullptr, 0, 0, lane);
  fc_layer<2>(V, 32, 1, sm.W6, 32, 30, sm.b6, true, SIGIN, 96, 36, nullptr, 0, 0, lane);
  fc_layer<2>(OD, 32, 1, sm.W7, 32, 30, sm.b7, true, SIN, 64, 9, nullptr, 0, 0, lane);

  // ---- two kgain steps (only second step's KGain is consumed) ----
  for (int step = 0; step < 2; ++step) {
    // Deep-prefetch tile 0 for this step's FC2 loop (overlaps with GRU phase).
    if (g.use_ws && wave == 0)
      tdm_load_1d(g.ws + WSOFF_T, streamOff0, REC_BYTES / 8, 3);

    gru<3>(F5, 32, 1, sm.WQih, 32, SIGIN, 96, 2, sm.WQhh, 64,
           sm.bQi, sm.bQh, 36, nullptr, 0, 0, lane);
    gru<3>(SIGIN, 96, 3, sm.WSigih, 96, FC2IN, 64, 2, sm.WSighh, 64,
           sm.bSi, sm.bSh, 36, FC4IN, 96, 0, lane);
    fc_layer<1>(FC2IN, 64, 2, sm.W1, 64, 9, sm.b1, true, SIN, 64, 0, nullptr, 0, 0, lane);
    gru<1>(SIN, 64, 2, sm.WSih, 64, F3IN, 32, 1, sm.WShh, 32,
           sm.bSSi, sm.bSSh, 9, FC2IN, 64, 36, lane);

    // ---- FC2a(45->1800, relu) fused with FC2b(1800->18), tiled by 32 hidden ----
    v8f acc2_0 = bias_init(sm.b2b, 18, 0, lane);
    v8f acc2_1 = bias_init(sm.b2b, 18, 1, lane);
    int buf = 0;
    for (int ht = 0; ht < HT_TILES; ++ht) {
      if (g.use_ws) {
        if (wave == 0) wait_tensor0();   // TDM for stream[buf] complete
        __syncthreads();                 // visible to all; stream[buf^1] free
        if (ht + 1 < HT_TILES) {         // overlap next tile's DMA with WMMAs
          if (wave == 0)
            tdm_load_1d(g.ws + WSOFF_T + (size_t)(ht + 1) * REC_BYTES,
                        streamOff0 + (unsigned)((buf ^ 1) * REC_BYTES),
                        REC_BYTES / 8, 3);
        }
        if (ht + 2 < HT_TILES)           // pull tile t+2 toward L2
          __builtin_prefetch(g.ws + WSOFF_T + (size_t)(ht + 2) * REC_BYTES + tid * 64,
                             0, 1);
      } else {
        __syncthreads();
        _Float16 *wa = (_Float16 *)&sm.stream[0][0];
        _Float16 *wb = (_Float16 *)&sm.stream[0][4096];
        float *fb = (float *)&sm.stream[0][6144];
        for (int i = tid; i < 32 * 64; i += NTHREADS) {
          int n = i >> 6, k = i & 63;
          int gn = ht * 32 + n;
          wa[i] = (_Float16)((gn < 1800 && k < 45) ? g.FC2a_W[gn * 45 + k] : 0.0f);
        }
        for (int i = tid; i < 32 * 32; i += NTHREADS) {
          int n = i >> 5, k = i & 31;
          int gk = ht * 32 + k;
          wb[i] = (_Float16)((n < 18 && gk < 1800) ? g.FC2b_W[n * 1800 + gk] : 0.0f);
        }
        if (tid < 32) {
          int gn = ht * 32 + tid;
          fb[tid] = (gn < 1800) ? g.FC2a_b[gn] : 0.0f;
        }
        __syncthreads();
        buf = 0;
      }
      const _Float16 *Wa = (const _Float16 *)&sm.stream[buf][0];
      const _Float16 *Wb = (const _Float16 *)&sm.stream[buf][4096];
      const float *b2a = (const float *)&sm.stream[buf][6144];
#pragma unroll
      for (int ot = 0; ot < 2; ++ot) {
        v8f h2 = wmma_gemm(FC2IN, 64, Wa, 64, 32, ot, 2,
                           bias_init(b2a, 32, ot, lane), lane);
        h2 = relu8(h2);
        storeC(H2A, 32, 0, ot, 32, h2, lane);          // h2 tile -> LDS as next A
      }
      asm volatile("" ::: "memory");                    // LDS in-order within wave
      acc2_0 = wmma_gemm(H2A, 32, Wb, 32, 32, 0, 1, acc2_0, lane);
      acc2_1 = wmma_gemm(H2A, 32, Wb, 32, 32, 1, 1, acc2_1, lane);
      if (g.use_ws) buf ^= 1;
    }
    __syncthreads();  // all waves done reading stream bufs before next step reuses them

    storeC(F3IN, 32, 9, 0, 18, acc2_0, lane);
    storeC(F3IN, 32, 9, 1, 18, acc2_1, lane);
    asm volatile("" ::: "memory");

    fc_layer<3>(F3IN, 32, 1, sm.W3, 32, 36, sm.b3, true, FC4IN, 96, 36,
                nullptr, 0, 0, lane);
    fc_layer<3>(FC4IN, 96, 3, sm.W4, 96, 36, sm.b4, true, FC2IN, 64, 0,
                FC4IN, 96, 0, lane);
  }

  // ---- Phase 3: INOV + make_state_valid, write output ----
  asm volatile("" ::: "memory");
  if (lane < 16) {
    int b = rowbase + lane;
    if (b < g.B) {
      float KG[18];
#pragma unroll
      for (int t = 0; t < 18; ++t) KG[t] = (float)F3IN[lane * 32 + 9 + t];
      float s[6];
#pragma unroll
      for (int i = 0; i < 6; ++i)
        s[i] = pr[i] + KG[i * 3 + 0] * dy0 + KG[i * 3 + 1] * dy1 + KG[i * 3 + 2] * dy2;
      float x  = fminf(fmaxf(s[0], 0.8f), 1.2f);
      float yv = fminf(fmaxf(s[1], 0.1f), 25.1f);
      float z  = -0.45f;
      float pn = sqrtf(x * x + yv * yv + z * z);
      float cc = fminf(fmaxf(x / pn, -1.0f), 1.0f);
      float ang = acosf(cc);
      const float CONE = 0.52359877559829887f;     // 30 deg
      const float COSC = 0.86602540378443865f;     // cos(30 deg)
      bool outside = ang > CONE;
      g.out[b * 6 + 0] = outside ? (x / COSC) : x;
      g.out[b * 6 + 1] = outside ? 0.0f : yv;
      g.out[b * 6 + 2] = outside ? 0.0f : z;
      g.out[b * 6 + 3] = s[3];
      g.out[b * 6 + 4] = s[4];
      g.out[b * 6 + 5] = s[5];
    }
  }
}

// ---------------- host launch ----------------
extern "C" void kernel_launch(void *const *d_in, const int *in_sizes, int n_in,
                              void *d_out, int out_size, void *d_ws, size_t ws_size,
                              hipStream_t stream) {
  (void)n_in; (void)out_size;
  KArgs g;
  g.y          = (const float *)d_in[0];
  g.post       = (const float *)d_in[1];
  g.post_prev  = (const float *)d_in[2];
  g.prior_prev = (const float *)d_in[3];
  g.y_prev     = (const float *)d_in[4];
  g.hQ0        = (const float *)d_in[5];
  g.hSig0      = (const float *)d_in[6];
  g.hS0        = (const float *)d_in[7];
  g.FC1_W  = (const float *)d_in[8];   g.FC1_b  = (const float *)d_in[9];
  g.FC2a_W = (const float *)d_in[10];  g.FC2a_b = (const float *)d_in[11];
  g.FC2b_W = (const float *)d_in[12];  g.FC2b_b = (const float *)d_in[13];
  g.FC3_W  = (const float *)d_in[14];  g.FC3_b  = (const float *)d_in[15];
  g.FC4_W  = (const float *)d_in[16];  g.FC4_b  = (const float *)d_in[17];
  g.FC5_W  = (const float *)d_in[18];  g.FC5_b  = (const float *)d_in[19];
  g.FC6_W  = (const float *)d_in[20];  g.FC6_b  = (const float *)d_in[21];
  g.FC7_W  = (const float *)d_in[22];  g.FC7_b  = (const float *)d_in[23];
  g.Q_Wih   = (const float *)d_in[24]; g.Q_Whh   = (const float *)d_in[25];
  g.Q_bih   = (const float *)d_in[26]; g.Q_bhh   = (const float *)d_in[27];
  g.Sig_Wih = (const float *)d_in[28]; g.Sig_Whh = (const float *)d_in[29];
  g.Sig_bih = (const float *)d_in[30]; g.Sig_bhh = (const float *)d_in[31];
  g.S_Wih   = (const float *)d_in[32]; g.S_Whh   = (const float *)d_in[33];
  g.S_bih   = (const float *)d_in[34]; g.S_bhh   = (const float *)d_in[35];
  g.ws = (const char *)d_ws;
  g.out = (float *)d_out;
  g.B = in_sizes[0] / 3;
  g.use_ws = (d_ws != nullptr && ws_size >= WS_NEEDED) ? 1 : 0;
  if (g.use_ws) {
    pack_all<<<dim3(HT_TILES + 1), dim3(256), 0, stream>>>(g, (char *)d_ws);
  }
  int blocks = (g.B + ROWS_PER_BLOCK - 1) / ROWS_PER_BLOCK;
  knet_fused<<<dim3(blocks), dim3(NTHREADS), 0, stream>>>(g);
}
